// S_Net_14147622273474
// MI455X (gfx1250) — matrix operand order repaired and checked by reading, not compiled, not run on hardware
//
#include <hip/hip_runtime.h>
#include <hip/hip_bf16.h>
#include <stdint.h>

#define N_NODES 100000
#define N_EDGES 3200000
#define F_IN    256
#define F_GCN   64
#define NTILES_M (N_NODES / 16)   // 6250 (exact)

typedef __attribute__((ext_vector_type(16))) _Float16 v16h;
typedef __attribute__((ext_vector_type(8)))  _Float16 v8h;
typedef __attribute__((ext_vector_type(8)))  float    v8f;

static __device__ __forceinline__ v8f wmma_f16(v16h a, v16h b, v8f c) {
    // D = A(16x32 f16) * B(32x16 f16) + C(16x16 f32)
    return __builtin_amdgcn_wmma_f32_16x16x32_f16(
        /*neg_a=*/false, a, /*neg_b=*/false, b,
        /*c_mod=*/(short)0, c, /*reuse_a=*/false, /*reuse_b=*/false);
}

// ---------------------------------------------------------------------------
// Graph preprocessing
// ---------------------------------------------------------------------------
__global__ void k_zero_u32(unsigned* p, int n) {
    int i = blockIdx.x * 256 + threadIdx.x;
    if (i < n) p[i] = 0u;
}

__global__ void k_count(const long long* __restrict__ ei, unsigned* __restrict__ deg) {
    int e = blockIdx.x * 256 + threadIdx.x;
    if (e >= N_EDGES) return;
    int dst = (int)ei[(size_t)N_EDGES + e];
    atomicAdd(&deg[dst], 1u);
}

__global__ void k_dinv(const unsigned* __restrict__ deg, float* __restrict__ dinv, int n) {
    int i = blockIdx.x * 256 + threadIdx.x;
    if (i < n) dinv[i] = rsqrtf((float)(deg[i] + 1u));   // +1 = self loop
}

// Single-block exclusive scan of in-degrees -> CSR row offsets (+ cursor copy)
__global__ void k_scan(const unsigned* __restrict__ deg, unsigned* __restrict__ rowstart,
                       unsigned* __restrict__ cursor, int n) {
    __shared__ unsigned sdata[1024];
    __shared__ unsigned sbase;
    if (threadIdx.x == 0) sbase = 0;
    __syncthreads();
    for (int base = 0; base < n; base += 1024) {
        int i = base + (int)threadIdx.x;
        unsigned v = (i < n) ? deg[i] : 0u;
        sdata[threadIdx.x] = v;
        __syncthreads();
        for (int off = 1; off < 1024; off <<= 1) {
            unsigned t = (threadIdx.x >= (unsigned)off) ? sdata[threadIdx.x - off] : 0u;
            __syncthreads();
            sdata[threadIdx.x] += t;
            __syncthreads();
        }
        unsigned excl = sdata[threadIdx.x] - v;
        if (i < n) {
            rowstart[i] = sbase + excl;
            cursor[i]   = sbase + excl;
        }
        __syncthreads();
        if (threadIdx.x == 1023) sbase += sdata[1023];
        __syncthreads();
    }
    if (threadIdx.x == 0) rowstart[n] = sbase;
}

__global__ void k_fill(const long long* __restrict__ ei, unsigned* __restrict__ cursor,
                       unsigned* __restrict__ colv) {
    int e = blockIdx.x * 256 + threadIdx.x;
    if (e >= N_EDGES) return;
    int src = (int)ei[e];
    int dst = (int)ei[(size_t)N_EDGES + e];
    unsigned pos = atomicAdd(&cursor[dst], 1u);
    colv[pos] = (unsigned)src;
}

// ---------------------------------------------------------------------------
// Pack fp32 weight [K x Ncols] (row-major, K = contraction dim) into the
// WMMA B-operand register layout, f16, tiled (Kp/32 x Np/16), zero-padded.
// Flat index: ((kt*ntiles + nt)*32 + lane)*16 + j  ;  K = kt*32 + (lane>>4)*8 + (j&7) + (j>>3)*16
// ---------------------------------------------------------------------------
__global__ void k_pack(const float* __restrict__ W, _Float16* __restrict__ dst,
                       int K, int Ncols, int Kp, int Np) {
    int t = blockIdx.x * 256 + threadIdx.x;
    int total = Kp * Np;
    if (t >= total) return;
    int j    = t & 15;
    int lane = (t >> 4) & 31;
    int tile = t >> 9;
    int ntiles = Np >> 4;
    int nt = tile % ntiles;
    int kt = tile / ntiles;
    int g  = lane >> 4;
    int n  = (lane & 15) + nt * 16;
    int k  = kt * 32 + g * 8 + (j & 7) + (j >> 3) * 16;
    float v = (k < K && n < Ncols) ? W[(size_t)k * Ncols + n] : 0.0f;
    dst[t] = (_Float16)v;
}

// ---------------------------------------------------------------------------
// h = x @ W_gcn   (100000x256 @ 256x64), one wave per 16-row tile, 4 n-tiles
// ---------------------------------------------------------------------------
__global__ void k_gemm(const float* __restrict__ x, const _Float16* __restrict__ wb,
                       float* __restrict__ h) {
    int gtid = blockIdx.x * (int)blockDim.x + (int)threadIdx.x;
    int wave = gtid >> 5;
    int lane = threadIdx.x & 31;
    if (wave >= NTILES_M) return;
    int m0  = wave * 16;
    int row = m0 + (lane & 15);
    int g   = lane >> 4;
    const float* xr = x + (size_t)row * F_IN + g * 8;

    v8f acc0 = {}, acc1 = {}, acc2 = {}, acc3 = {};
    #pragma unroll
    for (int kt = 0; kt < 8; ++kt) {
        const float* f = xr + kt * 32;
        float tmp[16];
        *(float4*)(tmp + 0)  = *(const float4*)(f + 0);
        *(float4*)(tmp + 4)  = *(const float4*)(f + 4);
        *(float4*)(tmp + 8)  = *(const float4*)(f + 16);
        *(float4*)(tmp + 12) = *(const float4*)(f + 20);
        v16h a;
        #pragma unroll
        for (int j = 0; j < 16; ++j) a[j] = (_Float16)tmp[j];

        v16h b0 = *(const v16h*)(wb + (size_t)(((kt * 4 + 0) * 32 + lane) * 16));
        v16h b1 = *(const v16h*)(wb + (size_t)(((kt * 4 + 1) * 32 + lane) * 16));
        v16h b2 = *(const v16h*)(wb + (size_t)(((kt * 4 + 2) * 32 + lane) * 16));
        v16h b3 = *(const v16h*)(wb + (size_t)(((kt * 4 + 3) * 32 + lane) * 16));
        acc0 = wmma_f16(a, b0, acc0);
        acc1 = wmma_f16(a, b1, acc1);
        acc2 = wmma_f16(a, b2, acc2);
        acc3 = wmma_f16(a, b3, acc3);
    }
    // C layout: element r -> row m0 + g*8 + r, col (lane&15) within each 16-tile
    float* hrow = h + (size_t)(m0 + g * 8) * F_GCN + (lane & 15);
    #pragma unroll
    for (int r = 0; r < 8; ++r) {
        hrow[(size_t)r * F_GCN + 0]  = acc0[r];
        hrow[(size_t)r * F_GCN + 16] = acc1[r];
        hrow[(size_t)r * F_GCN + 32] = acc2[r];
        hrow[(size_t)r * F_GCN + 48] = acc3[r];
    }
}

// ---------------------------------------------------------------------------
// CSR gather-aggregate:  hb[n] = relu( dinv[n]*(sum_e dinv[src]*h[src] + dinv[n]*h[n]) + b )
// 16 lanes per node, float4 per lane (64 cols)
// ---------------------------------------------------------------------------
__global__ void k_aggr(const float* __restrict__ h, const unsigned* __restrict__ rowstart,
                       const unsigned* __restrict__ colv, const float* __restrict__ dinv,
                       const float* __restrict__ bgcn, float* __restrict__ hb) {
    int tid  = blockIdx.x * 256 + (int)threadIdx.x;
    int node = tid >> 4;
    int c    = tid & 15;
    if (node >= N_NODES) return;
    unsigned s = rowstart[node], e = rowstart[node + 1];
    float ax = 0.f, ay = 0.f, az = 0.f, aw = 0.f;
    for (unsigned j = s; j < e; ++j) {
        unsigned src = colv[j];
        float sc = dinv[src];
        float4 hv = *(const float4*)(h + (size_t)src * F_GCN + c * 4);
        ax += sc * hv.x; ay += sc * hv.y; az += sc * hv.z; aw += sc * hv.w;
    }
    float dn = dinv[node];
    float4 hs = *(const float4*)(h + (size_t)node * F_GCN + c * 4);
    ax = dn * (ax + dn * hs.x);
    ay = dn * (ay + dn * hs.y);
    az = dn * (az + dn * hs.z);
    aw = dn * (aw + dn * hs.w);
    float4 bb = *(const float4*)(bgcn + c * 4);
    float4 o;
    o.x = fmaxf(ax + bb.x, 0.f);
    o.y = fmaxf(ay + bb.y, 0.f);
    o.z = fmaxf(az + bb.z, 0.f);
    o.w = fmaxf(aw + bb.w, 0.f);
    *(float4*)(hb + (size_t)node * F_GCN + c * 4) = o;
}

// ---------------------------------------------------------------------------
// Fused MLP: relu(hb@W1+b1) -> relu(@W2+b2) -> @W3+b3 ; wave per 16 rows.
// Layer outputs bounce through per-wave LDS (f16, row-major) to re-enter the
// WMMA A-operand layout. W3 tile is K-padded 16->32, N-padded 10->16.
// ---------------------------------------------------------------------------
__global__ void k_mlp(const float* __restrict__ hb,
                      const _Float16* __restrict__ wb1, const _Float16* __restrict__ wb2,
                      const _Float16* __restrict__ wb3,
                      const float* __restrict__ b1, const float* __restrict__ b2,
                      const float* __restrict__ b3, float* __restrict__ out) {
    __shared__ _Float16 smem[8][16 * 32];   // 1 KB per wave
    int w    = threadIdx.x >> 5;
    int lane = threadIdx.x & 31;
    int tile = blockIdx.x * 8 + w;
    if (tile >= NTILES_M) return;
    int m0  = tile * 16;
    int row = m0 + (lane & 15);
    int g   = lane >> 4;
    int col = lane & 15;
    _Float16* S = smem[w];

    // ---- layer 1: [16x64] @ [64x32] ----
    v8f acc0 = {}, acc1 = {};
    #pragma unroll
    for (int kt = 0; kt < 2; ++kt) {
        const float* f = hb + (size_t)row * F_GCN + kt * 32 + g * 8;
        float tmp[16];
        *(float4*)(tmp + 0)  = *(const float4*)(f + 0);
        *(float4*)(tmp + 4)  = *(const float4*)(f + 4);
        *(float4*)(tmp + 8)  = *(const float4*)(f + 16);
        *(float4*)(tmp + 12) = *(const float4*)(f + 20);
        v16h a;
        #pragma unroll
        for (int j = 0; j < 16; ++j) a[j] = (_Float16)tmp[j];
        v16h bb0 = *(const v16h*)(wb1 + (size_t)(((kt * 2 + 0) * 32 + lane) * 16));
        v16h bb1 = *(const v16h*)(wb1 + (size_t)(((kt * 2 + 1) * 32 + lane) * 16));
        acc0 = wmma_f16(a, bb0, acc0);
        acc1 = wmma_f16(a, bb1, acc1);
    }
    {
        float bia0 = b1[col];
        float bia1 = b1[16 + col];
        #pragma unroll
        for (int r = 0; r < 8; ++r) {
            int m = r + g * 8;
            float v0 = fmaxf(acc0[r] + bia0, 0.f);
            float v1 = fmaxf(acc1[r] + bia1, 0.f);
            S[m * 32 + col]      = (_Float16)v0;
            S[m * 32 + 16 + col] = (_Float16)v1;
        }
    }

    // ---- layer 2: [16x32] @ [32x16] ----  (A reloaded from LDS, wave-private)
    v16h a2;
    {
        v8h lo = *(const v8h*)(S + col * 32 + g * 8);
        v8h hi = *(const v8h*)(S + col * 32 + g * 8 + 16);
        #pragma unroll
        for (int j = 0; j < 8; ++j) { a2[j] = lo[j]; a2[8 + j] = hi[j]; }
    }
    v16h b2t = *(const v16h*)(wb2 + (size_t)(lane * 16));
    v8f zero = {};
    v8f c2 = wmma_f16(a2, b2t, zero);
    {
        float bia = b2[col];
        #pragma unroll
        for (int r = 0; r < 8; ++r) {
            int m = r + g * 8;
            float v = fmaxf(c2[r] + bia, 0.f);
            S[m * 32 + col]      = (_Float16)v;
            S[m * 32 + 16 + col] = (_Float16)0.f;   // K-pad for layer 3
        }
    }

    // ---- layer 3: [16x32(pad)] @ [32x16(pad)] -> 10 cols ----
    v16h a3;
    {
        v8h lo = *(const v8h*)(S + col * 32 + g * 8);
        v8h hi = *(const v8h*)(S + col * 32 + g * 8 + 16);
        #pragma unroll
        for (int j = 0; j < 8; ++j) { a3[j] = lo[j]; a3[8 + j] = hi[j]; }
    }
    v16h b3t = *(const v16h*)(wb3 + (size_t)(lane * 16));
    v8f c3 = wmma_f16(a3, b3t, zero);
    if (col < 10) {
        float bia = b3[col];
        #pragma unroll
        for (int r = 0; r < 8; ++r) {
            int m = m0 + r + g * 8;
            out[(size_t)m * 10 + col] = c3[r] + bia;
        }
    }
}

// ---------------------------------------------------------------------------
extern "C" void kernel_launch(void* const* d_in, const int* in_sizes, int n_in,
                              void* d_out, int out_size, void* d_ws, size_t ws_size,
                              hipStream_t stream) {
    const float*     x   = (const float*)d_in[0];
    const long long* ei  = (const long long*)d_in[1];
    const float*     Wg  = (const float*)d_in[2];
    const float*     bg  = (const float*)d_in[3];
    const float*     W1  = (const float*)d_in[4];
    const float*     b1  = (const float*)d_in[5];
    const float*     W2  = (const float*)d_in[6];
    const float*     b2  = (const float*)d_in[7];
    const float*     W3  = (const float*)d_in[8];
    const float*     b3  = (const float*)d_in[9];
    float* out = (float*)d_out;

    char*  ws  = (char*)d_ws;
    size_t off = 0;
    auto carve = [&](size_t bytes) -> void* {
        void* p = ws + off;
        off += (bytes + 255) & ~(size_t)255;
        return p;
    };
    _Float16* wb       = (_Float16*)carve(19456 * sizeof(_Float16)); // all packed weights
    float*    h        = (float*)   carve((size_t)N_NODES * F_GCN * 4);
    float*    hbuf     = (float*)   carve((size_t)N_NODES * F_GCN * 4);
    unsigned* deg      = (unsigned*)carve((size_t)N_NODES * 4);
    float*    dinv     = (float*)   carve((size_t)N_NODES * 4);
    unsigned* rowstart = (unsigned*)carve(((size_t)N_NODES + 1) * 4);
    unsigned* cursor   = (unsigned*)carve((size_t)N_NODES * 4);
    unsigned* colv     = (unsigned*)carve((size_t)N_EDGES * 4);

    _Float16* wbg = wb;                 // 256x64  -> 16384 halves
    _Float16* wb1 = wb + 16384;         // 64x32   -> 2048
    _Float16* wb2 = wb + 18432;         // 32x16   -> 512
    _Float16* wb3 = wb + 18944;         // 16x10 padded 32x16 -> 512

    int nb_nodes = (N_NODES + 255) / 256;
    int nb_edges = (N_EDGES + 255) / 256;
    int nb_waves = (NTILES_M * 32 + 255) / 256;   // 782 blocks, 8 waves each

    // Graph preprocessing
    k_zero_u32<<<nb_nodes, 256, 0, stream>>>(deg, N_NODES);
    k_count   <<<nb_edges, 256, 0, stream>>>(ei, deg);
    k_dinv    <<<nb_nodes, 256, 0, stream>>>(deg, dinv, N_NODES);
    k_scan    <<<1, 1024, 0, stream>>>(deg, rowstart, cursor, N_NODES);
    k_fill    <<<nb_edges, 256, 0, stream>>>(ei, cursor, colv);

    // Weight packing (f16, WMMA B-layout)
    k_pack<<<(256 * 64 + 255) / 256, 256, 0, stream>>>(Wg, wbg, 256, 64, 256, 64);
    k_pack<<<(64 * 32 + 255) / 256, 256, 0, stream>>>(W1, wb1, 64, 32, 64, 32);
    k_pack<<<(32 * 16 + 255) / 256, 256, 0, stream>>>(W2, wb2, 32, 16, 32, 16);
    k_pack<<<(32 * 16 + 255) / 256, 256, 0, stream>>>(W3, wb3, 16, 10, 32, 16);

    // h = x @ W_gcn (WMMA)
    k_gemm<<<nb_waves, 256, 0, stream>>>(x, wbg, h);

    // Normalized aggregation + bias + relu
    k_aggr<<<(N_NODES * 16 + 255) / 256, 256, 0, stream>>>(h, rowstart, colv, dinv, bg, hbuf);

    // Fused WMMA MLP -> d_out
    k_mlp<<<nb_waves, 256, 0, stream>>>(hbuf, wb1, wb2, wb3, b1, b2, b3, out);

    (void)in_sizes; (void)n_in; (void)out_size; (void)ws_size;
}
